// ScaledDotProductAttention_55482387529982
// MI455X (gfx1250) — matrix operand order repaired and checked by reading, not compiled, not run on hardware
//
#include <hip/hip_runtime.h>

// Scaled dot-product attention forward (causal), MI455X / gfx1250, wave32.
// Outputs: context [B,H,S,D] then attn_weights [B,H,S,S], concatenated in d_out.
// B=2 H=16 S=2048 D=64, fp32 in/out. scale = 1/sqrt(64) = 0.125.
//
// Bandwidth floor: 537MB attn write + 16MB ctx write + 48MB QKV read @ 23.3TB/s
// ~= 26us. Design: flash-style stats pass + single emit pass with NT stores;
// K/V tiles double-buffered in LDS via the Tensor Data Mover (TDM), WMMA f16
// for both GEMMs (Q pre-scaled by 0.125, exact in f16).

#define B_DIM 2
#define H_DIM 16
#define S_LEN 2048
#define D_DIM 64
#define NWAVE 4          // 4 waves per workgroup, 16 query rows each -> 64-row block
#define KSTRIDE 67       // f32 LDS row stride: 64 + 3 pad dwords (TDM pad), conflict-free

typedef __attribute__((ext_vector_type(16))) _Float16 v16h;
typedef __attribute__((ext_vector_type(8)))  float    v8f;
typedef __attribute__((ext_vector_type(4)))  unsigned int u32x4;
typedef __attribute__((ext_vector_type(8)))  int      i32x8;
typedef __attribute__((ext_vector_type(4)))  int      i32x4;

#if defined(__has_builtin)
#if __has_builtin(__builtin_amdgcn_tensor_load_to_lds)
#define USE_TDM 1
#endif
#endif
#ifndef USE_TDM
#define USE_TDM 0
#endif

union HF { v16h v; _Float16 h[16]; };
union FV { v8f  v; float    f[8];  };

__device__ __forceinline__ float hmax16(float x) {
    x = fmaxf(x, __shfl_xor(x, 1, 32));
    x = fmaxf(x, __shfl_xor(x, 2, 32));
    x = fmaxf(x, __shfl_xor(x, 4, 32));
    x = fmaxf(x, __shfl_xor(x, 8, 32));
    return x;
}
__device__ __forceinline__ float hsum16(float x) {
    x += __shfl_xor(x, 1, 32);
    x += __shfl_xor(x, 2, 32);
    x += __shfl_xor(x, 4, 32);
    x += __shfl_xor(x, 8, 32);
    return x;
}

#if USE_TDM
// Issue a TDM 2D tile load: rows x 64 f32 from global (row stride 64 elems) into
// LDS with +3-dword padding per 64-dword row (-> KSTRIDE=67 f32 LDS stride).
// D# layout per CDNA5 ISA ch.10.8 (group0: count/lds/global/type, group1: dims).
// This toolchain exposes the 6-arg builtin:
//   (u32x4 g0, i32x8 g1, i32x4 g2, i32x4 g3, i32x8 extra, i32 cpol)
__device__ __forceinline__ void tdm_load_tile(const float* gsrc, float* lds_dst,
                                              int rows) {
    unsigned long long ga = (unsigned long long)(uintptr_t)gsrc;
    unsigned lds = (unsigned)(uintptr_t)lds_dst;       // low 32 bits = LDS byte addr
    u32x4 g0;
    g0[0] = 1u;                                        // count=1, load, user mode
    g0[1] = lds;                                       // lds_addr (bytes)
    g0[2] = (unsigned)ga;                              // global_addr[31:0]
    g0[3] = (unsigned)((ga >> 32) & 0x1FFFFFFu)        // global_addr[56:32]
          | (2u << 30);                                // type = 2 (image/tensor)
    i32x8 g1;
    g1[0] = (int)((2u << 16)                           // data_size = 4B
                | (1u << 20)                           // pad_enable
                | (5u << 22)                           // pad_interval: 64 dwords
                | (2u << 25));                         // pad_amount: 3 dwords
    g1[1] = (int)(64u << 16);                          // tensor_dim0 = 64 (low16)
    g1[2] = (int)(((unsigned)S_LEN) << 16);            // dim0 hi=0 | tensor_dim1 lo16
    g1[3] = (int)(64u << 16);                          // dim1 hi=0 | tile_dim0 = 64
    g1[4] = rows & 0xFFFF;                             // tile_dim1 = rows, tile_dim2=0
    g1[5] = 64;                                        // tensor_dim0_stride = 64 elems
    g1[6] = 0;
    g1[7] = 0;
    i32x4 z4 = {0, 0, 0, 0};                           // groups 2/3 unused (2D tile)
    i32x8 z8 = {0, 0, 0, 0, 0, 0, 0, 0};
    __builtin_amdgcn_tensor_load_to_lds(g0, g1, z4, z4, z8, 0);
}
#endif

__global__ __launch_bounds__(NWAVE * 32)
void fa_fwd_kernel(const float* __restrict__ Qg, const float* __restrict__ Kg,
                   const float* __restrict__ Vg, float* __restrict__ ctx,
                   float* __restrict__ attn)
{
    __shared__ float    sK[2][32 * KSTRIDE];          // double-buffered K tiles (f32)
    __shared__ float    sV[2][32 * KSTRIDE];          // double-buffered V tiles (f32)
    __shared__ _Float16 sP[NWAVE * 16 * 34];          // per-wave 16x32 prob bounce

    const int tid  = threadIdx.x;
    const int w    = tid >> 5;
    const int lane = tid & 31;
    const int hi   = lane >> 4;
    const int ln   = lane & 15;

    const int bh     = blockIdx.y;
    const int qblock = blockIdx.x * 64;
    const int qr0    = qblock + w * 16;

    const float* Qb = Qg + (size_t)bh * S_LEN * D_DIM;
    const float* Kb = Kg + (size_t)bh * S_LEN * D_DIM;
    const float* Vb = Vg + (size_t)bh * S_LEN * D_DIM;
    float*       Cb = ctx  + (size_t)bh * S_LEN * D_DIM;
    float*       Ab = attn + (size_t)bh * S_LEN * S_LEN;

    // ---- Q tile as two 16x32 f16 A-fragments, pre-scaled by 0.125 ----
    // A operand k order used here (B matches): lane holds row M=lane%16; element h:
    //   K = h + 16*(lane>=16)
    HF a0, a1;
    #pragma unroll
    for (int h = 0; h < 16; ++h) {
        const int v  = h >> 1;
        const int kk = h + 16 * hi; (void)v;
        a0.h[h] = (_Float16)(0.125f * Qb[(qr0 + ln) * D_DIM + kk]);
        a1.h[h] = (_Float16)(0.125f * Qb[(qr0 + ln) * D_DIM + 32 + kk]);
    }

    const int jTop = (qblock >> 4) + 3;               // last causal key tile for WG

    float mrow[8], lrow[8];
    #pragma unroll
    for (int r = 0; r < 8; ++r) { mrow[r] = -3.0e38f; lrow[r] = 0.0f; }

    // ======================= Phase 1: streaming softmax stats =======================
#if USE_TDM
    if (w == 0) tdm_load_tile(Kb, &sK[0][0], 16);     // prologue: tile 0 in flight
#endif
    for (int j = 0; j <= jTop; ++j) {
#if USE_TDM
        if (w == 0) {
            if (j < jTop) {
                tdm_load_tile(Kb + (size_t)(j + 1) * 16 * D_DIM, &sK[(j + 1) & 1][0], 16);
                __builtin_amdgcn_s_wait_tensorcnt(1); // tile j landed
            } else {
                __builtin_amdgcn_s_wait_tensorcnt(0);
            }
        }
        __syncthreads();                              // tile j visible to all waves
#else
        __syncthreads();
        #pragma unroll
        for (int e = 0; e < 8; ++e) {
            const int idx = tid + 128 * e;
            const int row = idx >> 6, col = idx & 63;
            sK[j & 1][row * KSTRIDE + col] = Kb[(j * 16 + row) * D_DIM + col];
        }
        __syncthreads();
#endif
        const float* kb = &sK[j & 1][0];

        if (j * 16 <= qr0 + 15) {                     // wave-uniform causal skip
            HF b0, b1;                                // B: N=lane%16, K=h+16*(lane>=16)
            #pragma unroll
            for (int h = 0; h < 16; ++h) {
                const int kl = h + 16 * hi;
                b0.h[h] = (_Float16)kb[ln * KSTRIDE + kl];
                b1.h[h] = (_Float16)kb[ln * KSTRIDE + 32 + kl];
            }
            FV c;
            #pragma unroll
            for (int i = 0; i < 8; ++i) c.f[i] = 0.0f;
            c.v = __builtin_amdgcn_wmma_f32_16x16x32_f16(false, a0.v, false, b0.v,
                                                         (short)0, c.v, false, false);
            c.v = __builtin_amdgcn_wmma_f32_16x16x32_f16(false, a1.v, false, b1.v,
                                                         (short)0, c.v, false, false);
            #pragma unroll
            for (int r = 0; r < 8; ++r) {
                const int q = qr0 + r + 8 * hi;
                const int k = j * 16 + ln;
                const bool valid = (k <= q);
                const float s  = valid ? c.f[r] : -3.0e38f;
                const float nm = fmaxf(mrow[r], hmax16(s));
                const float e1 = valid ? __expf(s - nm) : 0.0f;
                const float es = hsum16(e1);
                lrow[r] = lrow[r] * __expf(mrow[r] - nm) + es;
                mrow[r] = nm;
            }
        }
        __syncthreads();                              // buf (j&1) free for reuse
    }

    float invl[8];
    #pragma unroll
    for (int r = 0; r < 8; ++r) invl[r] = 1.0f / lrow[r];

    // ============== Phase 2: emit probs (NT stores) + accumulate P*V ==============
    FV acc[4];
    #pragma unroll
    for (int d = 0; d < 4; ++d)
        #pragma unroll
        for (int i = 0; i < 8; ++i) acc[d].f[i] = 0.0f;

    const int nPair = (jTop + 1) >> 1;                // 32-key steps
    _Float16* myP = sP + w * 16 * 34;

#if USE_TDM
    if (w == 0) {                                     // prologue: pair 0 in flight
        tdm_load_tile(Kb, &sK[0][0], 32);
        tdm_load_tile(Vb, &sV[0][0], 32);
    }
#endif
    for (int jp = 0; jp < nPair; ++jp) {
#if USE_TDM
        if (w == 0) {
            if (jp + 1 < nPair) {
                tdm_load_tile(Kb + (size_t)(jp + 1) * 32 * D_DIM, &sK[(jp + 1) & 1][0], 32);
                tdm_load_tile(Vb + (size_t)(jp + 1) * 32 * D_DIM, &sV[(jp + 1) & 1][0], 32);
                __builtin_amdgcn_s_wait_tensorcnt(2); // pair jp landed
            } else {
                __builtin_amdgcn_s_wait_tensorcnt(0);
            }
        }
        __syncthreads();
#else
        __syncthreads();
        #pragma unroll
        for (int e = 0; e < 16; ++e) {
            const int idx = tid + 128 * e;
            const int row = idx >> 6, col = idx & 63;
            sK[jp & 1][row * KSTRIDE + col] = Kb[(jp * 32 + row) * D_DIM + col];
            sV[jp & 1][row * KSTRIDE + col] = Vb[(jp * 32 + row) * D_DIM + col];
        }
        __syncthreads();
#endif
        const float* kb = &sK[jp & 1][0];
        const float* vb = &sV[jp & 1][0];

        #pragma unroll
        for (int t = 0; t < 2; ++t) {                 // two 16-key score tiles
            const int j = jp * 2 + t;
            HF b0, b1;
            #pragma unroll
            for (int h = 0; h < 16; ++h) {
                const int kl = h + 16 * hi;
                b0.h[h] = (_Float16)kb[(16 * t + ln) * KSTRIDE + kl];
                b1.h[h] = (_Float16)kb[(16 * t + ln) * KSTRIDE + 32 + kl];
            }
            FV c;
            #pragma unroll
            for (int i = 0; i < 8; ++i) c.f[i] = 0.0f;
            c.v = __builtin_amdgcn_wmma_f32_16x16x32_f16(false, a0.v, false, b0.v,
                                                         (short)0, c.v, false, false);
            c.v = __builtin_amdgcn_wmma_f32_16x16x32_f16(false, a1.v, false, b1.v,
                                                         (short)0, c.v, false, false);
            #pragma unroll
            for (int r = 0; r < 8; ++r) {
                const int q = qr0 + r + 8 * hi;
                const int k = j * 16 + ln;
                const bool valid = (k <= q);
                const float p = valid ? __expf(c.f[r] - mrow[r]) * invl[r] : 0.0f;
                __builtin_nontemporal_store(p, &Ab[(size_t)q * S_LEN + k]);
                myP[(r + 8 * hi) * 34 + 16 * t + ln] = (_Float16)p;
            }
        }
        __syncthreads();                              // publish sP across lanes

        HF pa;                                        // P 16x32 as A-fragment
        #pragma unroll
        for (int h = 0; h < 16; ++h) {
            const int v  = h >> 1;
            const int kk = h + 16 * hi; (void)v;
            pa.h[h] = myP[ln * 34 + kk];
        }
        #pragma unroll
        for (int dd = 0; dd < 4; ++dd) {              // V 32x16 B-fragments per d-slab
            HF bv;
            #pragma unroll
            for (int h = 0; h < 16; ++h) {
                const int kl = h + 16 * hi;
                bv.h[h] = (_Float16)vb[kl * KSTRIDE + dd * 16 + ln];
            }
            acc[dd].v = __builtin_amdgcn_wmma_f32_16x16x32_f16(false, pa.v, false, bv.v,
                                                               (short)0, acc[dd].v,
                                                               false, false);
        }
        __syncthreads();                              // buf (jp&1) free for reuse
    }

    // ---- Write context (C layout: M = r + 8*hi, N = lane%16) ----
    #pragma unroll
    for (int dd = 0; dd < 4; ++dd)
        #pragma unroll
        for (int r = 0; r < 8; ++r)
            Cb[(size_t)(qr0 + r + 8 * hi) * D_DIM + dd * 16 + ln] = acc[dd].f[r];

    // ---- Zero-fill attn tail beyond the causal block (cols >= qblock+64) ----
    const int kTail = 32 * nPair;                     // == qblock + 64
    for (int r = 0; r < 16; ++r) {
        float* rowp = Ab + (size_t)(qr0 + r) * S_LEN;
        for (int k = kTail + lane; k < S_LEN; k += 32)
            __builtin_nontemporal_store(0.0f, rowp + k);
    }
}

extern "C" void kernel_launch(void* const* d_in, const int* in_sizes, int n_in,
                              void* d_out, int out_size, void* d_ws, size_t ws_size,
                              hipStream_t stream) {
    (void)in_sizes; (void)n_in; (void)out_size; (void)d_ws; (void)ws_size;
    const float* Q = (const float*)d_in[0];
    const float* K = (const float*)d_in[1];
    const float* V = (const float*)d_in[2];
    // d_in[3] is the causal mask (int32); causality is applied analytically.
    float* out  = (float*)d_out;
    float* ctx  = out;                                            // [B,H,S,D]
    float* attn = out + (size_t)B_DIM * H_DIM * S_LEN * D_DIM;    // [B,H,S,S]

    dim3 grid(S_LEN / 64, B_DIM * H_DIM);
    dim3 block(NWAVE * 32);
    fa_fwd_kernel<<<grid, block, 0, stream>>>(Q, K, V, ctx, attn);
}
